// Retrieval2D_53558242181150
// MI455X (gfx1250) — matrix operand order repaired and checked by reading, not compiled, not run on hardware
//
#include <hip/hip_runtime.h>

// ---------------- types ----------------
typedef __attribute__((ext_vector_type(16))) __bf16 v16bf;
typedef __attribute__((ext_vector_type(8)))  __bf16 v8bf;
typedef __attribute__((ext_vector_type(8)))  float  v8f;

#define D_DIM   2048
#define B_DIM   256
#define L_DIM   30
#define TSTRIDE 2052   // t_feat row stride (D+4)
#define TM      256    // all query rows in one block -> db streamed from HBM exactly once
#define TN      128
#define TK      32
#define NTHR    512    // 16 wave32 waves

// order-preserving float<->uint transform
__device__ __forceinline__ unsigned int ford(float f) {
  unsigned int u = __float_as_uint(f);
  return (u & 0x80000000u) ? ~u : (u | 0x80000000u);
}
__device__ __forceinline__ float forddec(unsigned int o) {
  unsigned int u = (o & 0x80000000u) ? (o & 0x7fffffffu) : ~o;
  return __uint_as_float(u);
}

// ---------------- kernel 1: query prep ----------------
__global__ __launch_bounds__(256) void qprep_kernel(const float* __restrict__ t_feat,
                                                    __bf16* __restrict__ qbf,
                                                    float* __restrict__ inv_q,
                                                    unsigned long long* __restrict__ keys) {
  __shared__ float red[256];
  const int b = blockIdx.x;
  const int t = threadIdx.x;
  const float4* row = (const float4*)(t_feat + (size_t)b * TSTRIDE); // 2052*4 % 16 == 0
  float s = 0.f;
  #pragma unroll
  for (int it = 0; it < 2; ++it) {
    int i4 = t * 2 + it;
    float4 f = row[i4];
    s += f.x * f.x + f.y * f.y + f.z * f.z + f.w * f.w;
    __bf16* dst = qbf + (size_t)b * D_DIM + i4 * 4;
    dst[0] = (__bf16)f.x; dst[1] = (__bf16)f.y; dst[2] = (__bf16)f.z; dst[3] = (__bf16)f.w;
  }
  red[t] = s;
  __syncthreads();
  for (int off = 128; off > 0; off >>= 1) {
    if (t < off) red[t] += red[t + off];
    __syncthreads();
  }
  if (t == 0) {
    inv_q[b] = rsqrtf(red[0]);
    keys[b]  = 0ull;
  }
}

// ---------------- kernel 2: fused bf16 WMMA GEMM + norm + tile argmax ----------------
__global__ __launch_bounds__(NTHR) void gemm_argmax_kernel(const float* __restrict__ db,
                                                           const __bf16* __restrict__ qbf,
                                                           const float* __restrict__ inv_q,
                                                           unsigned long long* __restrict__ keys,
                                                           int N) {
  // Phase-overlapped LDS arena:
  //   loop phase : As (TM*TK bf16 = 16 KB) | Bs (TN*TK bf16 = 8 KB)
  //   epilogue   : Cs (TM*TN f32 = 128 KB) | Bn4 (TN*4 f32 = 2 KB) | invb (TN f32)
  constexpr int AS_BYTES = TM * TK * 2;                        // 16384
  constexpr int CS_BYTES = TM * TN * 4;                        // 131072
  constexpr int SM_BYTES = CS_BYTES + TN * 4 * 4 + TN * 4;     // 133632
  __shared__ __align__(16) char smem[SM_BYTES];
  __bf16 (*As)[TK]  = (__bf16(*)[TK])smem;
  __bf16 (*Bs)[TK]  = (__bf16(*)[TK])(smem + AS_BYTES);
  float  (*Cs)[TN]  = (float(*)[TN])smem;
  float  (*Bn4)[4]  = (float(*)[4])(smem + CS_BYTES);
  float  *invb      = (float*)(smem + CS_BYTES + TN * 4 * 4);

  const int t    = threadIdx.x;
  const int lane = t & 31;
  const int wave = t >> 5;          // 0..15
  const int mw   = wave >> 1;       // 0..7  -> rows mw*32 .. +32
  const int nw   = wave & 1;        // 0..1  -> cols nw*64 .. +64
  const int nBase = blockIdx.x * TN;

  const int lrow  = t >> 1;         // 0..255 A-tile row this thread loads
  const int lhalf = t & 1;
  const int r = lane & 15;
  const int h = lane >> 4;

  v8f acc[2][4] = {};
  float bnorm = 0.f;

  // B tile spread across ALL 512 threads: 8 floats (32 B) of db per thread per k-step
  const int  brow    = t >> 2;               // 0..127
  const int  bq      = t & 3;                // 8-float quarter within the 32-float K chunk
  const int  n_this  = nBase + brow;
  const bool n_valid = (n_this < N);
  const float* dbrow = db + (size_t)n_this * D_DIM + bq * 8;

  for (int k0 = 0; k0 < D_DIM; k0 += TK) {
    // ---- A tile: bf16 queries (resident in L2) ----
    {
      const __bf16* src = qbf + (size_t)lrow * D_DIM + k0 + lhalf * 16;
      v8bf v0 = *(const v8bf*)(src);
      v8bf v1 = *(const v8bf*)(src + 8);
      *(v8bf*)&As[lrow][lhalf * 16]     = v0;
      *(v8bf*)&As[lrow][lhalf * 16 + 8] = v1;
    }
    // ---- B tile: fp32 db -> bf16, fused ||db_n||^2 accumulation ----
    {
      __bf16 tmp[8];
      if (n_valid) {
        const float4* src = (const float4*)(dbrow + k0);
        if (k0 + TK < D_DIM) __builtin_prefetch(dbrow + k0 + TK, 0, 0);
        #pragma unroll
        for (int q4 = 0; q4 < 2; ++q4) {
          float4 f = src[q4];
          bnorm += f.x * f.x + f.y * f.y + f.z * f.z + f.w * f.w;
          tmp[q4 * 4 + 0] = (__bf16)f.x; tmp[q4 * 4 + 1] = (__bf16)f.y;
          tmp[q4 * 4 + 2] = (__bf16)f.z; tmp[q4 * 4 + 3] = (__bf16)f.w;
        }
      } else {
        #pragma unroll
        for (int i = 0; i < 8; ++i) tmp[i] = (__bf16)0.f;
      }
      *(v8bf*)&Bs[brow][bq * 8] = *(v8bf*)&tmp[0];   // one 16 B ds_store
    }
    __syncthreads();

    // ---- fragments ----
    // A 16-bit 16x32: elems0-7 = K[h*8 .. +8], elems8-15 = K[16+h*8 .. +8]
    v16bf afrag[2];
    #pragma unroll
    for (int mt = 0; mt < 2; ++mt) {
      int row = mw * 32 + mt * 16 + r;
      union { v16bf v; v8bf half[2]; } u;
      u.half[0] = *(const v8bf*)&As[row][h * 8];
      u.half[1] = *(const v8bf*)&As[row][16 + h * 8];
      afrag[mt] = u.v;
    }
    // B 16-bit 32x16: elems0-7 = K[h*8 .. +8], elems8-15 = K[16+h*8 .. +8]
    v16bf bfrag[4];
    #pragma unroll
    for (int nt = 0; nt < 4; ++nt) {
      int col = nw * 64 + nt * 16 + r;
      union { v16bf v; v8bf half[2]; } u;
      u.half[0] = *(const v8bf*)&Bs[col][h * 8];
      u.half[1] = *(const v8bf*)&Bs[col][16 + h * 8];
      bfrag[nt] = u.v;
    }
    #pragma unroll
    for (int mt = 0; mt < 2; ++mt)
      #pragma unroll
      for (int nt = 0; nt < 4; ++nt)
        acc[mt][nt] = __builtin_amdgcn_wmma_f32_16x16x32_bf16(
            false, afrag[mt], false, bfrag[nt], (short)0, acc[mt][nt], false, false);
    __syncthreads();
  }

  // ---- epilogue: norms, score tile, per-row argmax ----
  Bn4[brow][bq] = bnorm;            // Bn4 region does not overlap As/Bs

  // D layout: lanes 0-15: m = vgpr, n = lane; lanes 16-31: m = 8+vgpr, n = lane-16
  #pragma unroll
  for (int mt = 0; mt < 2; ++mt)
    #pragma unroll
    for (int nt = 0; nt < 4; ++nt)
      #pragma unroll
      for (int e = 0; e < 8; ++e) {
        int m = mw * 32 + mt * 16 + h * 8 + e;
        int n = nw * 64 + nt * 16 + r;
        Cs[m][n] = acc[mt][nt][e];
      }
  __syncthreads();

  if (t < TN) invb[t] = rsqrtf(Bn4[t][0] + Bn4[t][1] + Bn4[t][2] + Bn4[t][3]);
  __syncthreads();

  if (t < TM) {
    const float iq = inv_q[t];
    float best = -3.0e38f;
    int bestn = nBase;
    for (int j = 0; j < TN; ++j) {
      int n = nBase + j;
      if (n < N) {
        float s = Cs[t][j] * iq * invb[j];
        if (s > best) { best = s; bestn = n; }
      }
    }
    unsigned long long key =
        ((unsigned long long)ford(best) << 32) | (unsigned int)(~(unsigned int)bestn);
    atomicMax(keys + t, key);
  }
}

// ---------------- kernel 3: decode + gather captions ----------------
__global__ __launch_bounds__(256) void gather_kernel(const unsigned long long* __restrict__ keys,
                                                     const int* __restrict__ lang_ids,
                                                     float* __restrict__ out, int N) {
  int b = threadIdx.x + blockIdx.x * blockDim.x;
  if (b >= B_DIM) return;
  unsigned long long key = keys[b];
  unsigned int idx = ~(unsigned int)(key & 0xffffffffu);
  float score = forddec((unsigned int)(key >> 32));
  if (idx >= (unsigned int)N) idx = 0;
  const int* cap = lang_ids + (size_t)idx * L_DIM;
  #pragma unroll
  for (int j = 0; j < L_DIM; ++j)
    out[(size_t)b * L_DIM + j] = (float)cap[j];
  out[(size_t)B_DIM * L_DIM + b] = score;
}

// ---------------- launch ----------------
extern "C" void kernel_launch(void* const* d_in, const int* in_sizes, int n_in,
                              void* d_out, int out_size, void* d_ws, size_t ws_size,
                              hipStream_t stream) {
  const float* t_feat   = (const float*)d_in[0];
  const float* db       = (const float*)d_in[1];
  const int*   lang_ids = (const int*)d_in[2];
  const int N = in_sizes[1] / D_DIM;   // 50000

  char* ws = (char*)d_ws;
  size_t off = 0;
  float* inv_q = (float*)(ws + off);                 off += (size_t)B_DIM * 4;
  off = (off + 255) & ~(size_t)255;
  unsigned long long* keys = (unsigned long long*)(ws + off); off += (size_t)B_DIM * 8;
  off = (off + 255) & ~(size_t)255;
  __bf16* qbf = (__bf16*)(ws + off);                 // 1 MB

  qprep_kernel<<<B_DIM, 256, 0, stream>>>(t_feat, qbf, inv_q, keys);

  dim3 grid((N + TN - 1) / TN);                      // 391 workgroups, db streamed once
  gemm_argmax_kernel<<<grid, NTHR, 0, stream>>>(db, qbf, inv_q, keys, N);

  gather_kernel<<<1, 256, 0, stream>>>(keys, lang_ids, (float*)d_out, N);
}
